// CompressiveEncoder_55611236549043
// MI455X (gfx1250) — matrix operand order repaired and checked by reading, not compiled, not run on hardware
//
#include <hip/hip_runtime.h>
#include <hip/hip_bf16.h>
#include <math.h>

// Model constants
#define B_     4
#define SEQ_   1024
#define D_     512
#define H_     8
#define DH_    64
#define MEM_   1024
#define CMEM_  256
#define KV_    2304   // CMEM + MEM + SEQ
#define FF_    2048
#define L_     4
#define RATIO_ 4

typedef __attribute__((ext_vector_type(16))) __bf16 v16bf;
typedef __attribute__((ext_vector_type(8)))  float  v8f;

__device__ __forceinline__ __bf16 f2bf(float f) {
  union { float f; unsigned u; } v; v.f = f;
  unsigned r = (v.u + 0x7FFFu + ((v.u >> 16) & 1u)) >> 16;
  union { unsigned short s; __bf16 b; } o; o.s = (unsigned short)r;
  return o.b;
}
__device__ __forceinline__ float bf2f(__bf16 h) {
  union { unsigned short s; __bf16 b; } i; i.b = h;
  union { unsigned u; float f; } o; o.u = (unsigned)i.s << 16;
  return o.f;
}
__device__ __forceinline__ float gelu_exact(float t) {
  return 0.5f * t * (1.0f + erff(t * 0.70710678118654752f));
}

// One-shot async 16B copy global -> LDS (tracked by ASYNCcnt).
__device__ __forceinline__ void async_cp_b128(void* lds_ptr, const void* gptr) {
  unsigned lds = (unsigned)(unsigned long long)(size_t)lds_ptr;  // low 32b = LDS offset
  asm volatile("global_load_async_to_lds_b128 %0, %1, off"
               :: "v"(lds), "v"((unsigned long long)(size_t)gptr)
               : "memory");
}
__device__ __forceinline__ void wait_asynccnt0() {
  asm volatile("s_wait_asynccnt 0x0" ::: "memory");
}

// ---------------------------------------------------------------------------
// Batched-strided WMMA GEMM. A: (M,K) bf16 (ABF) or f32 (converted at staging).
// B: bf16, NN (K,N) or NT (N,K). C: f32 or bf16 (OUTBF).
// Tile 128x64x32, 256 threads / 8 waves, wave = 32x32 via 2x2
// V_WMMA_F32_16X16X32_BF16. bf16 LDS tiles, double buffered.
// Pure-copy staging paths use GLOBAL_LOAD_ASYNC_TO_LDS_B128 (+ asynccnt fence);
// converting/transposing paths use manual b128 load + ds store.
// All dims assumed M%128==N%64==K%32==0.
// ---------------------------------------------------------------------------
enum { EPI_SCALE = 0, EPI_BIAS = 1, EPI_BIAS_GELU = 2, EPI_BIAS_RESID = 3 };

template <int EPI, bool TB, bool ABF, bool OUTBF>
__global__ __launch_bounds__(256) void gemm_wmma(
    const void* __restrict__ Av, const __bf16* __restrict__ Bmv,
    void* __restrict__ Cv, const float* __restrict__ bias,
    int M, int N, int K, int lda, int ldb, int ldc,
    long long sA, long long sB, long long sC, float alpha)
{
  __shared__ __align__(16) __bf16 As[2][128][40];  // pitch 80B (16B aligned)
  __shared__ __align__(16) __bf16 Bt[2][64][40];   // B stored transposed: [n][k]

  const int tid = threadIdx.x;
  const int z = blockIdx.z;
  const int rowBase = blockIdx.y * 128;
  const int colBase = blockIdx.x * 64;
  const int wid = tid >> 5, lane = tid & 31;
  const int wm = wid & 3, wn = wid >> 2;          // 4x2 wave grid
  const int lh = lane >> 4, lm = lane & 15;

  const __bf16* Agh = (const __bf16*)Av + (long long)z * sA;  // valid if ABF
  const float*  Agf = (const float*)Av + (long long)z * sA;   // valid if !ABF
  const __bf16* Bg  = Bmv + (long long)z * sB;

  auto stage = [&](int buf, int k0) {
    if (ABF) {
#pragma unroll
      for (int i = 0; i < 2; ++i) {               // 512 async b128 = 128x32 bf16
        int idx = i * 256 + tid;
        int r = idx >> 2, c8 = idx & 3;
        async_cp_b128(&As[buf][r][c8 * 8],
                      Agh + (long long)(rowBase + r) * lda + k0 + c8 * 8);
      }
    } else {
#pragma unroll
      for (int i = 0; i < 4; ++i) {               // 1024 float4 = 128x32 f32
        int idx = i * 256 + tid;
        int r = idx >> 3, c4 = idx & 7;
        float4 f = *(const float4*)(Agf + (long long)(rowBase + r) * lda + k0 + c4 * 4);
        union { uint2 u; __bf16 h[4]; } o;
        o.h[0] = f2bf(f.x); o.h[1] = f2bf(f.y); o.h[2] = f2bf(f.z); o.h[3] = f2bf(f.w);
        *(uint2*)(&As[buf][r][c4 * 4]) = o.u;
      }
    }
    if (TB) {                                     // B (N,K): contiguous k async copy
      int n = tid >> 2, c8 = tid & 3;
      async_cp_b128(&Bt[buf][n][c8 * 8],
                    Bg + (long long)(colBase + n) * ldb + k0 + c8 * 8);
    } else {                                      // B (K,N): transpose scatter
      int k = tid >> 3, n8 = tid & 7;
      union { uint4 q; __bf16 h[8]; } u;
      u.q = *(const uint4*)(Bg + (long long)(k0 + k) * ldb + colBase + n8 * 8);
#pragma unroll
      for (int j = 0; j < 8; ++j) Bt[buf][n8 * 8 + j][k] = u.h[j];
    }
  };
  auto fence_stage = [&]() {
    if (ABF || TB) wait_asynccnt0();              // our asm isn't compiler-tracked
  };

  v8f acc[2][2];
#pragma unroll
  for (int i = 0; i < 2; ++i)
#pragma unroll
    for (int j = 0; j < 2; ++j)
#pragma unroll
      for (int e = 0; e < 8; ++e) acc[i][j][e] = 0.f;

  const int nk = K >> 5;
  stage(0, 0);
  fence_stage();
  __syncthreads();

  union FragU { v16bf v; uint4 q[2]; };

  for (int kt = 0; kt < nk; ++kt) {
    const int cur = kt & 1;
    if (kt + 1 < nk) stage(cur ^ 1, (kt + 1) << 5);
    if (kt + 2 < nk) {
      const int pk = (kt + 2) << 5;
      if (ABF) __builtin_prefetch(Agh + (long long)(rowBase + (tid >> 1)) * lda + pk, 0, 0);
      else     __builtin_prefetch(Agf + (long long)(rowBase + (tid >> 1)) * lda + pk, 0, 0);
      if (TB)  __builtin_prefetch(Bg + (long long)(colBase + (tid & 63)) * ldb + pk, 0, 0);
      else     __builtin_prefetch(Bg + (long long)(pk + (tid & 31)) * ldb + colBase, 0, 0);
    }

    // Fragments (ISA 7.12.2). A 16x32: elems 0..7 = K kb..kb+7, 8..15 = kb+16..kb+23
// (kb = lh*8) -> two contiguous 16B chunks. B 32x16 via transposed LDS:
    // same K order as A (elems 0..7 = kb..kb+7, 8..15 = kb+16..kb+23).
    FragU fa[2], fb[2];
#pragma unroll
    for (int t2 = 0; t2 < 2; ++t2) {
      const __bf16* Ar = &As[cur][wm * 32 + t2 * 16 + lm][0];
      fa[t2].q[0] = *(const uint4*)(Ar + lh * 8);
      fa[t2].q[1] = *(const uint4*)(Ar + lh * 8 + 16);
      const __bf16* Br = &Bt[cur][wn * 32 + t2 * 16 + lm][0];
      fb[t2].q[0] = *(const uint4*)(Br + lh * 8);
      fb[t2].q[1] = *(const uint4*)(Br + lh * 8 + 16);
    }
#pragma unroll
    for (int i = 0; i < 2; ++i)
#pragma unroll
      for (int j = 0; j < 2; ++j)
        acc[i][j] = __builtin_amdgcn_wmma_f32_16x16x32_bf16(
            false, fa[i].v, false, fb[j].v, (short)0, acc[i][j], false, false);

    if (kt + 1 < nk) fence_stage();   // staged buffer complete before publish
    __syncthreads();
  }

  // Epilogue. C/D: VGPR r -> M = r + lh*8, N = lm (per 16x16 tile)
  float*  Cf = (float*)Cv;
  __bf16* Ch = (__bf16*)Cv;
  const long long cb = (long long)z * sC;
#pragma unroll
  for (int i = 0; i < 2; ++i) {
#pragma unroll
    for (int j = 0; j < 2; ++j) {
      const int tr = rowBase + wm * 32 + i * 16 + lh * 8;
      const int tc = colBase + wn * 32 + j * 16 + lm;
#pragma unroll
      for (int r = 0; r < 8; ++r) {
        long long off = cb + (long long)(tr + r) * ldc + tc;
        float v = acc[i][j][r];
        float res;
        if (EPI == EPI_SCALE)          res = alpha * v;
        else if (EPI == EPI_BIAS)      res = v + bias[tc];
        else if (EPI == EPI_BIAS_GELU) res = gelu_exact(v + bias[tc]);
        else {
          float prev = OUTBF ? bf2f(Ch[off]) : Cf[off];
          res = prev + v + bias[tc];
        }
        if (OUTBF) Ch[off] = f2bf(res); else Cf[off] = res;
      }
    }
  }
}

// ---------------------------------------------------------------------------
// Helper kernels
// ---------------------------------------------------------------------------
__global__ __launch_bounds__(256) void k_cvt(const float* __restrict__ s,
                                             __bf16* __restrict__ d, long long n) {
  long long i = ((long long)blockIdx.x * 256 + threadIdx.x) * 4;
  if (i >= n) return;
  float4 f = *(const float4*)(s + i);
  union { uint2 u; __bf16 h[4]; } o;
  o.h[0] = f2bf(f.x); o.h[1] = f2bf(f.y); o.h[2] = f2bf(f.z); o.h[3] = f2bf(f.w);
  *(uint2*)(d + i) = o.u;
}

__global__ __launch_bounds__(256) void k_embed(const int* __restrict__ seq,
                                               const float* __restrict__ embed,
                                               float* __restrict__ x) {
  int row = blockIdx.x;
  const float* e = embed + (long long)seq[row] * D_;
  float* o = x + (long long)row * D_;
  for (int c = threadIdx.x; c < D_; c += 256) o[c] = e[c];
}

// LayerNorm. yf (f32, optional) and yb (bf16) outputs.
__global__ __launch_bounds__(256) void k_layernorm(const float* __restrict__ x,
                                                   const float* __restrict__ g,
                                                   const float* __restrict__ bt,
                                                   float* __restrict__ yf,
                                                   __bf16* __restrict__ yb) {
  __shared__ float red[256];
  int row = blockIdx.x, t = threadIdx.x;
  const float* xr = x + (long long)row * D_;
  float v0 = xr[t], v1 = xr[t + 256];
  red[t] = v0 + v1;
  __syncthreads();
  for (int s = 128; s > 0; s >>= 1) { if (t < s) red[t] += red[t + s]; __syncthreads(); }
  float mu = red[0] * (1.0f / D_);
  __syncthreads();
  float d0 = v0 - mu, d1 = v1 - mu;
  red[t] = d0 * d0 + d1 * d1;
  __syncthreads();
  for (int s = 128; s > 0; s >>= 1) { if (t < s) red[t] += red[t + s]; __syncthreads(); }
  float inv = rsqrtf(red[0] * (1.0f / D_) + 1e-5f);
  float r0 = d0 * inv * g[t] + bt[t];
  float r1 = d1 * inv * g[t + 256] + bt[t + 256];
  long long o = (long long)row * D_ + t;
  if (yf) { yf[o] = r0; yf[o + 256] = r1; }
  yb[o] = f2bf(r0); yb[o + 256] = f2bf(r1);
}

__global__ __launch_bounds__(256) void k_concat_kv(const float* __restrict__ cmem,
                                                   const float* __restrict__ mem,
                                                   const float* __restrict__ xn,
                                                   __bf16* __restrict__ kv) {
  long long i = (long long)blockIdx.x * 256 + threadIdx.x;
  if (i >= (long long)B_ * KV_ * D_) return;
  int c = (int)(i % D_);
  long long tt = i / D_;
  int pos = (int)(tt % KV_);
  int b   = (int)(tt / KV_);
  float v;
  if (pos < CMEM_)              v = cmem[((long long)b * CMEM_ + pos) * D_ + c];
  else if (pos < CMEM_ + MEM_)  v = mem[((long long)b * MEM_ + (pos - CMEM_)) * D_ + c];
  else                          v = xn[((long long)b * SEQ_ + (pos - CMEM_ - MEM_)) * D_ + c];
  kv[i] = f2bf(v);
}

// dots += shift(pos_dots); mask; softmax (in place, f32). (H,SEQ,KV) per batch.
__global__ __launch_bounds__(256) void k_attn_softmax(float* __restrict__ dots,
                                                      const float* __restrict__ posd,
                                                      const unsigned char* __restrict__ mask,
                                                      int b) {
  __shared__ float red[256];
  int i = blockIdx.x, h = blockIdx.y, t = threadIdx.x;
  float* row = dots + ((long long)h * SEQ_ + i) * KV_;
  const float* pr = posd + ((long long)h * SEQ_ + i) * KV_;
  bool mi = mask[b * SEQ_ + i] != 0;
  float mx = -3.402823466e38f;
  for (int j = t; j < KV_; j += 256) {
    float v = row[j];
    int jj = 1023 + j - i;                  // _shift rel-position gather
    if (jj < KV_) v += pr[jj];
    bool valid = (j < KV_ - SEQ_) || (mi && mask[b * SEQ_ + (j - (KV_ - SEQ_))] != 0);
    if (!valid) v = -3.402823466e38f;
    row[j] = v;
    mx = fmaxf(mx, v);
  }
  red[t] = mx; __syncthreads();
  for (int s = 128; s > 0; s >>= 1) { if (t < s) red[t] = fmaxf(red[t], red[t + s]); __syncthreads(); }
  mx = red[0]; __syncthreads();
  float sum = 0.f;
  for (int j = t; j < KV_; j += 256) { float e = __expf(row[j] - mx); row[j] = e; sum += e; }
  red[t] = sum; __syncthreads();
  for (int s = 128; s > 0; s >>= 1) { if (t < s) red[t] += red[t + s]; __syncthreads(); }
  float inv = 1.0f / red[0];
  for (int j = t; j < KV_; j += 256) row[j] *= inv;
}

__global__ __launch_bounds__(256) void k_softmax_rows(float* __restrict__ buf, int ncols) {
  __shared__ float red[256];
  int i = blockIdx.x, h = blockIdx.y, t = threadIdx.x;
  float* row = buf + ((long long)h * SEQ_ + i) * (long long)ncols;
  float mx = -3.402823466e38f;
  for (int j = t; j < ncols; j += 256) mx = fmaxf(mx, row[j]);
  red[t] = mx; __syncthreads();
  for (int s = 128; s > 0; s >>= 1) { if (t < s) red[t] = fmaxf(red[t], red[t + s]); __syncthreads(); }
  mx = red[0]; __syncthreads();
  float sum = 0.f;
  for (int j = t; j < ncols; j += 256) { float e = __expf(row[j] - mx); row[j] = e; sum += e; }
  red[t] = sum; __syncthreads();
  for (int s = 128; s > 0; s >>= 1) { if (t < s) red[t] += red[t + s]; __syncthreads(); }
  float inv = 1.0f / red[0];
  for (int j = t; j < ncols; j += 256) row[j] *= inv;
}

__global__ __launch_bounds__(256) void k_wacc(const float* __restrict__ probs,
                                              float* __restrict__ wacc) {
  long long idx = (long long)blockIdx.x * 256 + threadIdx.x;
  if (idx >= (long long)SEQ_ * KV_) return;
  float s = 0.f;
#pragma unroll
  for (int h = 0; h < H_; ++h) s += probs[(long long)h * SEQ_ * KV_ + idx];
  wacc[idx] += s * (1.0f / (B_ * H_ * L_));
}

// A_conv[(b*256+t), i*4+r] = mem[b, t*4+r, i]  (conv as GEMM), bf16 out.
__global__ __launch_bounds__(256) void k_pack_conv(const float* __restrict__ mem,
                                                   __bf16* __restrict__ Ac) {
  long long idx = (long long)blockIdx.x * 256 + threadIdx.x;
  if (idx >= (long long)B_ * CMEM_ * (D_ * RATIO_)) return;
  int kk = (int)(idx % (D_ * RATIO_));
  long long t = idx / (D_ * RATIO_);
  int tt = (int)(t % CMEM_);
  int b  = (int)(t / CMEM_);
  int i = kk >> 2, r = kk & 3;
  Ac[idx] = f2bf(mem[((long long)b * MEM_ + tt * RATIO_ + r) * D_ + i]);
}

__global__ __launch_bounds__(256) void k_aux_reduce(const float* __restrict__ o1,
                                                    const float* __restrict__ o2,
                                                    float* __restrict__ aux) {
  __shared__ float red[256];
  int t = threadIdx.x;
  long long n = (long long)SEQ_ * D_;
  float s = 0.f;
  for (long long idx = (long long)blockIdx.x * 256 + t; idx < n; idx += 256ll * gridDim.x) {
    float d = o1[idx] - o2[idx];
    s += d * d;
  }
  red[t] = s; __syncthreads();
  for (int st = 128; st > 0; st >>= 1) { if (t < st) red[t] += red[t + st]; __syncthreads(); }
  if (t == 0) atomicAdd(aux, red[0] * (1.0f / ((float)B_ * H_ * SEQ_ * DH_ * L_)));
}

// ---------------------------------------------------------------------------
// Host side
// ---------------------------------------------------------------------------
template <int EPI, bool TB, bool ABF, bool OUTBF>
static inline void G(const void* A, const __bf16* B, void* C, const float* bias,
                     int M, int N, int K, int lda, int ldb, int ldc,
                     long long sA, long long sB, long long sC,
                     int batch, float alpha, hipStream_t s) {
  dim3 g(N / 64, M / 128, batch), blk(256);
  gemm_wmma<EPI, TB, ABF, OUTBF><<<g, blk, 0, s>>>(A, B, C, bias, M, N, K,
                                                   lda, ldb, ldc, sA, sB, sC, alpha);
}

static inline char* carve(char*& p, size_t bytes) {
  char* r = p;
  p += (bytes + 255) & ~(size_t)255;
  return r;
}

extern "C" void kernel_launch(void* const* d_in, const int* in_sizes, int n_in,
                              void* d_out, int out_size, void* d_ws, size_t ws_size,
                              hipStream_t stream) {
  (void)in_sizes; (void)n_in; (void)out_size; (void)ws_size;

  const int*           seq     = (const int*)d_in[0];
  const unsigned char* mask    = (const unsigned char*)d_in[1];
  const float*         mems    = (const float*)d_in[2];
  const float*         cmems   = (const float*)d_in[3];
  const float*         pos_emb = (const float*)d_in[4];
  const float*         embed   = (const float*)d_in[5];
  const float*         ln1_g   = (const float*)d_in[6];
  const float*         ln1_b   = (const float*)d_in[7];
  const float*         Wq      = (const float*)d_in[8];
  const float*         Wkv     = (const float*)d_in[9];
  const float*         Wo      = (const float*)d_in[10];
  const float*         bo      = (const float*)d_in[11];
  const float*         conv_w  = (const float*)d_in[12];
  const float*         conv_b  = (const float*)d_in[13];
  const float*         ln2_g   = (const float*)d_in[14];
  const float*         ln2_b   = (const float*)d_in[15];
  const float*         W1      = (const float*)d_in[16];
  const float*         b1      = (const float*)d_in[17];
  const float*         W2      = (const float*)d_in[18];
  const float*         b2      = (const float*)d_in[19];

  // Output: x | new_mems | new_cmems | aux | w_acc
  float* out       = (float*)d_out;
  float* out_x     = out;
  float* out_mems  = out_x + (long long)B_ * SEQ_ * D_;
  float* out_cmems = out_mems + (long long)L_ * B_ * MEM_ * D_;
  float* out_aux   = out_cmems + (long long)L_ * B_ * CMEM_ * D_;
  float* out_wacc  = out_aux + 1;

  // Workspace carve
  char* p = (char*)d_ws;
  float*  x       = (float*)carve(p, sizeof(float) * (size_t)B_ * SEQ_ * D_);
  float*  dots    = (float*)carve(p, sizeof(float) * (size_t)H_ * SEQ_ * KV_);
  float*  posd    = (float*)carve(p, sizeof(float) * (size_t)H_ * SEQ_ * KV_);
  float*  auxd1   = (float*)carve(p, sizeof(float) * (size_t)H_ * SEQ_ * MEM_);
  float*  auxd2   = (float*)carve(p, sizeof(float) * (size_t)H_ * SEQ_ * CMEM_);
  float*  outok   = (float*)carve(p, sizeof(float) * (size_t)SEQ_ * D_);
  float*  outck   = (float*)carve(p, sizeof(float) * (size_t)SEQ_ * D_);
  __bf16* xn_bf   = (__bf16*)carve(p, 2 * (size_t)B_ * SEQ_ * D_);
  __bf16* q_bf    = (__bf16*)carve(p, 2 * (size_t)B_ * SEQ_ * D_);
  __bf16* kvin_bf = (__bf16*)carve(p, 2 * (size_t)B_ * KV_ * D_);
  __bf16* kvo_bf  = (__bf16*)carve(p, 2 * (size_t)B_ * KV_ * 2 * D_);
  __bf16* atto_bf = (__bf16*)carve(p, 2 * (size_t)B_ * SEQ_ * D_);
  __bf16* xn2_bf  = (__bf16*)carve(p, 2 * (size_t)B_ * SEQ_ * D_);
  __bf16* ffh_bf  = (__bf16*)carve(p, 2 * (size_t)B_ * SEQ_ * FF_);
  __bf16* cvA_bf  = (__bf16*)carve(p, 2 * (size_t)B_ * CMEM_ * D_ * RATIO_);
  __bf16* ckcv_bf = (__bf16*)carve(p, 2 * (size_t)B_ * CMEM_ * 2 * D_);
  __bf16* pos_bf  = (__bf16*)carve(p, 2 * (size_t)H_ * KV_ * DH_);
  __bf16* Wq_bf   = (__bf16*)carve(p, 2 * (size_t)D_ * D_);
  __bf16* Wkv_bf  = (__bf16*)carve(p, 2 * (size_t)D_ * 2 * D_);
  __bf16* Wo_bf   = (__bf16*)carve(p, 2 * (size_t)D_ * D_);
  __bf16* cw_bf   = (__bf16*)carve(p, 2 * (size_t)D_ * D_ * RATIO_);
  __bf16* W1_bf   = (__bf16*)carve(p, 2 * (size_t)D_ * FF_);
  __bf16* W2_bf   = (__bf16*)carve(p, 2 * (size_t)FF_ * D_);

  hipMemsetAsync(out_aux, 0, (1 + (size_t)SEQ_ * KV_) * sizeof(float), stream);

  auto cvt = [&](const float* s, __bf16* d, long long n) {
    k_cvt<<<(int)((n / 4 + 255) / 256), 256, 0, stream>>>(s, d, n);
  };

  k_embed<<<B_ * SEQ_, 256, 0, stream>>>(seq, embed, x);
  cvt(pos_emb, pos_bf, (long long)H_ * KV_ * DH_);

  const float scale = 0.125f;  // DH^-0.5

  for (int l = 0; l < L_; ++l) {
    const float* mem_l  = mems  + (long long)l * B_ * MEM_ * D_;
    const float* cmem_l = cmems + (long long)l * B_ * CMEM_ * D_;
    float* xn_out = out_mems  + (long long)l * B_ * MEM_ * D_;   // new_mems[l]
    float* comp   = out_cmems + (long long)l * B_ * CMEM_ * D_;  // new_cmems[l]

    // Per-layer weight conversion to bf16
    cvt(Wq + (long long)l * D_ * D_, Wq_bf, (long long)D_ * D_);
    cvt(Wkv + (long long)l * D_ * 2 * D_, Wkv_bf, (long long)D_ * 2 * D_);
    cvt(Wo + (long long)l * D_ * D_, Wo_bf, (long long)D_ * D_);
    cvt(conv_w + (long long)l * D_ * D_ * RATIO_, cw_bf, (long long)D_ * D_ * RATIO_);
    cvt(W1 + (long long)l * D_ * FF_, W1_bf, (long long)D_ * FF_);
    cvt(W2 + (long long)l * FF_ * D_, W2_bf, (long long)FF_ * D_);

    // LN1 -> f32 new_mems output + bf16 shadow
    k_layernorm<<<B_ * SEQ_, 256, 0, stream>>>(x, ln1_g + l * D_, ln1_b + l * D_, xn_out, xn_bf);

    // q = xn @ Wq
    G<EPI_SCALE, false, true, true>(xn_bf, Wq_bf, q_bf, nullptr,
        B_ * SEQ_, D_, D_, D_, D_, D_, 0, 0, 0, 1, 1.0f, stream);

    // kv_in = concat(cmem, mem, xn) (bf16)
    {
      long long tot = (long long)B_ * KV_ * D_;
      k_concat_kv<<<(int)((tot + 255) / 256), 256, 0, stream>>>(cmem_l, mem_l, xn_out, kvin_bf);
    }
    // kvout = kv_in @ Wkv
    G<EPI_SCALE, false, true, true>(kvin_bf, Wkv_bf, kvo_bf, nullptr,
        B_ * KV_, 2 * D_, D_, D_, 2 * D_, 2 * D_, 0, 0, 0, 1, 1.0f, stream);

    // ---- Attention (per batch, heads via gridDim.z) ----
    for (int b = 0; b < B_; ++b) {
      const __bf16* qb = q_bf + (long long)b * SEQ_ * D_;
      const __bf16* kb = kvo_bf + (long long)b * KV_ * 2 * D_;
      const __bf16* vb = kb + D_;

      G<EPI_SCALE, true, true, false>(qb, kb, dots, nullptr,
          SEQ_, KV_, DH_, D_, 2 * D_, KV_,
          DH_, DH_, (long long)SEQ_ * KV_, H_, scale, stream);
      G<EPI_SCALE, true, true, false>(qb, pos_bf, posd, nullptr,
          SEQ_, KV_, DH_, D_, DH_, KV_,
          DH_, (long long)KV_ * DH_, (long long)SEQ_ * KV_, H_, scale, stream);

      { dim3 g(SEQ_, H_); k_attn_softmax<<<g, 256, 0, stream>>>(dots, posd, mask, b); }
      {
        long long tot = (long long)SEQ_ * KV_;
        k_wacc<<<(int)((tot + 255) / 256), 256, 0, stream>>>(dots, out_wacc);
      }
      // attnout = probs(f32) @ v  -> bf16 output
      G<EPI_SCALE, false, false, true>(dots, vb, atto_bf + (long long)b * SEQ_ * D_, nullptr,
          SEQ_, DH_, KV_, KV_, 2 * D_, D_,
          (long long)SEQ_ * KV_, DH_, DH_, H_, 1.0f, stream);
    }

    // x = x + attnout @ Wo + bo
    G<EPI_BIAS_RESID, false, true, false>(atto_bf, Wo_bf, x, bo + (long long)l * D_,
        B_ * SEQ_, D_, D_, D_, D_, D_, 0, 0, 0, 1, 1.0f, stream);

    // ---- Compression conv (old_mem == mems[l]) ----
    {
      long long tot = (long long)B_ * CMEM_ * (D_ * RATIO_);
      k_pack_conv<<<(int)((tot + 255) / 256), 256, 0, stream>>>(mem_l, cvA_bf);
    }
    G<EPI_BIAS, true, true, false>(cvA_bf, cw_bf, comp, conv_b + (long long)l * D_,
        B_ * CMEM_, D_, D_ * RATIO_, D_ * RATIO_, D_ * RATIO_, D_, 0, 0, 0, 1, 1.0f, stream);

    // ckcv = comp(f32) @ Wkv -> bf16
    G<EPI_SCALE, false, false, true>(comp, Wkv_bf, ckcv_bf, nullptr,
        B_ * CMEM_, 2 * D_, D_, D_, 2 * D_, 2 * D_, 0, 0, 0, 1, 1.0f, stream);

    // ---- Aux loss ----
    for (int b = 0; b < B_; ++b) {
      const __bf16* qb  = q_bf + (long long)b * SEQ_ * D_;
      const __bf16* okb = kvo_bf + (long long)b * KV_ * 2 * D_ + (long long)CMEM_ * 2 * D_;
      const __bf16* ovb = okb + D_;
      const __bf16* ckb = ckcv_bf + (long long)b * CMEM_ * 2 * D_;
      const __bf16* cvb = ckb + D_;

      G<EPI_SCALE, true, true, false>(qb, okb, auxd1, nullptr,
          SEQ_, MEM_, DH_, D_, 2 * D_, MEM_,
          DH_, DH_, (long long)SEQ_ * MEM_, H_, scale, stream);
      { dim3 g(SEQ_, H_); k_softmax_rows<<<g, 256, 0, stream>>>(auxd1, MEM_); }
      G<EPI_SCALE, false, false, false>(auxd1, ovb, outok, nullptr,
          SEQ_, DH_, MEM_, MEM_, 2 * D_, D_,
          (long long)SEQ_ * MEM_, DH_, DH_, H_, 1.0f, stream);

      G<EPI_SCALE, true, true, false>(qb, ckb, auxd2, nullptr,
          SEQ_, CMEM_, DH_, D_, 2 * D_, CMEM_,
          DH_, DH_, (long long)SEQ_ * CMEM_, H_, scale, stream);
      { dim3 g(SEQ_, H_); k_softmax_rows<<<g, 256, 0, stream>>>(auxd2, CMEM_); }
      G<EPI_SCALE, false, false, false>(auxd2, cvb, outck, nullptr,
          SEQ_, DH_, CMEM_, CMEM_, 2 * D_, D_,
          (long long)SEQ_ * CMEM_, DH_, DH_, H_, 1.0f, stream);

      k_aux_reduce<<<256, 256, 0, stream>>>(outok, outck, out_aux);
    }

    // ---- FFN ----
    k_layernorm<<<B_ * SEQ_, 256, 0, stream>>>(x, ln2_g + l * D_, ln2_b + l * D_, nullptr, xn2_bf);
    G<EPI_BIAS_GELU, false, true, true>(xn2_bf, W1_bf, ffh_bf, b1 + (long long)l * FF_,
        B_ * SEQ_, FF_, D_, D_, FF_, FF_, 0, 0, 0, 1, 1.0f, stream);
    G<EPI_BIAS_RESID, false, true, false>(ffh_bf, W2_bf, x, b2 + (long long)l * D_,
        B_ * SEQ_, D_, FF_, FF_, D_, D_, 0, 0, 0, 1, 1.0f, stream);
  }

  hipMemcpyAsync(out_x, x, (size_t)B_ * SEQ_ * D_ * sizeof(float),
                 hipMemcpyDeviceToDevice, stream);
}
